// LinearBase_44573170598240
// MI455X (gfx1250) — matrix operand order repaired and checked by reading, not compiled, not run on hardware
//
#include <hip/hip_runtime.h>

typedef __attribute__((ext_vector_type(16))) _Float16 v16h;
typedef __attribute__((ext_vector_type(8)))  _Float16 v8h;
typedef __attribute__((ext_vector_type(2)))  _Float16 h2v;
typedef __attribute__((ext_vector_type(8)))  float    v8f;

#define IN_F      256
#define OUT_F     256
#define NB        11                    // GRID_SIZE + SPLINE_ORDER
#define K_SPLINE  (NB * IN_F)           // 2816
#define K_TOTAL   (K_SPLINE + IN_F)     // 3072
#define M_TILE    64
#define K_CHUNK   32
#define N_CHUNKS  (K_TOTAL / K_CHUNK)   // 96
#define N_SPLINE_CHUNKS (K_SPLINE / K_CHUNK) // 88
#define A_BUF_ELEMS (M_TILE * K_CHUNK)  // 2048 halves per A chunk buffer

union V16U { v16h v; v8h h[2]; };

// ---------------------------------------------------------------------------
// Kernel 1: pack [coeff (O,I,NB) | base_w (O,I)] -> Wt[col=o][k] f16,
// k = n*256 + i (spline), k = 2816 + i (SiLU). k-contiguous per column so a
// WMMA B-fragment is two contiguous global_load_b128 per lane.
// ---------------------------------------------------------------------------
__global__ __launch_bounds__(256)
void pack_weights_kernel(const float* __restrict__ coeff,
                         const float* __restrict__ base_w,
                         _Float16* __restrict__ Wt) {
    int idx = blockIdx.x * 256 + threadIdx.x;     // idx = o*3072 + k
    if (idx >= OUT_F * K_TOTAL) return;
    int o = idx / K_TOTAL;
    int k = idx - o * K_TOTAL;
    float v;
    if (k < K_SPLINE) {
        int n = k >> 8;
        int i = k & 255;
        v = coeff[(o * IN_F + i) * NB + n];
    } else {
        v = base_w[o * IN_F + (k - K_SPLINE)];
    }
    Wt[idx] = (_Float16)v;
}

// Cardinal cubic B-spline N3(u), support [0,4), via truncated powers:
//   N3(u) = (1/6) * sum_{j=0..4} (-1)^j C(4,j) max(u-j,0)^3
// Branchless: max/mul/fma only -> dual-issues and co-executes with WMMA.
__device__ __forceinline__ float bspline3(float u) {
    float p0 = fmaxf(u,       0.f);
    float p1 = fmaxf(u - 1.f, 0.f);
    float p2 = fmaxf(u - 2.f, 0.f);
    float p3 = fmaxf(u - 3.f, 0.f);
    float p4 = fmaxf(u - 4.f, 0.f);
    float c0 = p0 * p0 * p0;
    float c1 = p1 * p1 * p1;
    float c2 = p2 * p2 * p2;
    float c3 = p3 * p3 * p3;
    float c4 = p4 * p4 * p4;
    return (c0 + c4 + 6.f * c2 - 4.f * (c1 + c3)) * (1.f / 6.f);
}

// Build one 64x32 f16 basis chunk (chunk index kc < 88) into dst.
// Thread t handles column pair 2*(t&15), rows (t>>4)*4 .. +3:
// ds_load_b64 of z pairs, packed ds_store_b32 of two halves.
__device__ __forceinline__ void buildChunk(int kc, const float* zbuf,
                                           _Float16* dst, int tid) {
    const int cp = tid & 15;
    const int rg = tid >> 4;
    const int n  = kc >> 3;            // 8 chunks per basis block
    const int i0 = (kc & 7) * K_CHUNK;
    const float sh = (float)(3 - n);
    #pragma unroll
    for (int r = 0; r < 4; ++r) {
        const int m = rg * 4 + r;
        const float2 z2 = *(const float2*)(zbuf + m * IN_F + i0 + 2 * cp);
        h2v v;
        v.x = (_Float16)bspline3(fmaf(z2.x + 1.f, 4.f, sh));  // u=(z-t_n)/h
        v.y = (_Float16)bspline3(fmaf(z2.y + 1.f, 4.f, sh));
        *(h2v*)(dst + m * K_CHUNK + 2 * cp) = v;
    }
}

// ---------------------------------------------------------------------------
// Kernel 2: fused LayerNorm -> basis/SiLU expansion -> f16 WMMA GEMM.
// 256 threads (8 wave32), 64 rows x 256 cols per workgroup.
// Wave w owns cols [32w, 32w+32): 4 M-tiles x 2 N-tiles accumulators.
// Single barrier per K-chunk: A chunk double-buffered in LDS, B fragments
// double-buffered in VGPRs (loads for k+1 issue before WMMAs of k).
// ---------------------------------------------------------------------------
__global__ __launch_bounds__(256)
void kan_fused_kernel(const float* __restrict__ x,
                      const float* __restrict__ gamma,
                      const float* __restrict__ beta,
                      const _Float16* __restrict__ Wt,
                      float* __restrict__ out) {
    extern __shared__ char smem_raw[];
    float*    zbuf = (float*)smem_raw;                    // [64][256] f32 (64 KB)
    _Float16* sbuf = (_Float16*)(zbuf + M_TILE * IN_F);   // [64][256] f16 (32 KB)
    _Float16* abuf = sbuf + M_TILE * IN_F;                // 2 x [64][32] f16 (8 KB)

    const int tid  = threadIdx.x;
    const int wave = tid >> 5;
    const int lane = tid & 31;
    const int g    = lane >> 4;                 // lane half (0/1)
    const int rowBase = blockIdx.x * M_TILE;

    // ---- Phase 1: LayerNorm (wave32 shuffle reduce) + SiLU, into LDS ----
    for (int r = 0; r < 8; ++r) {
        const int row = wave * 8 + r;
        const float* xr = x + (size_t)(rowBase + row) * IN_F;
        float xv[8], s = 0.f, s2 = 0.f;
        #pragma unroll
        for (int c = 0; c < 8; ++c) {
            xv[c] = xr[lane + 32 * c];
            s  += xv[c];
            s2 += xv[c] * xv[c];
        }
        #pragma unroll
        for (int m = 16; m >= 1; m >>= 1) {
            s  += __shfl_xor(s,  m, 32);
            s2 += __shfl_xor(s2, m, 32);
        }
        const float mu   = s * (1.f / 256.f);
        const float var  = s2 * (1.f / 256.f) - mu * mu;
        const float rstd = rsqrtf(var + 1e-5f);
        #pragma unroll
        for (int c = 0; c < 8; ++c) {
            const int i = lane + 32 * c;
            zbuf[row * IN_F + i] = (xv[c] - mu) * rstd * gamma[i] + beta[i];
            sbuf[row * IN_F + i] = (_Float16)(xv[c] / (1.f + __expf(-xv[c])));
        }
    }
    __syncthreads();                 // zbuf/sbuf visible to all waves

    v8f acc[4][2];
    #pragma unroll
    for (int mt = 0; mt < 4; ++mt)
        #pragma unroll
        for (int nt = 0; nt < 2; ++nt)
            acc[mt][nt] = (v8f){};

    // B fragment loads: 32x16 f16, lane half holds K = j + 16g, col = lane&15
#define LOADB(kc_, dstArr)                                                    \
    do {                                                                      \
        const int kk_ = (kc_) * K_CHUNK;                                      \
        _Pragma("unroll")                                                     \
        for (int nt = 0; nt < 2; ++nt) {                                      \
            const int col_ = wave * 32 + nt * 16 + (lane & 15);               \
            const _Float16* bp_ = Wt + (size_t)col_ * K_TOTAL + kk_ + 16 * g; \
            dstArr[nt].h[0] = *(const v8h*)(bp_);                             \
            dstArr[nt].h[1] = *(const v8h*)(bp_ + 8);                         \
        }                                                                     \
    } while (0)

    // One pipeline step: consume chunk kb with B regs BC, preload B for kb+1
    // into BN, and build basis chunk kb+1 into the alternate LDS buffer.
#define STEP(kb_, BC, BN)                                                     \
    do {                                                                      \
        const int kbv_ = (kb_);                                               \
        const int k0_  = kbv_ * K_CHUNK;                                      \
        __syncthreads();                                                      \
        LOADB((kbv_ + 1 < N_CHUNKS) ? (kbv_ + 1) : kbv_, BN);                 \
        if (kbv_ + 1 < N_SPLINE_CHUNKS)                                       \
            buildChunk(kbv_ + 1, zbuf,                                        \
                       abuf + ((kbv_ + 1) & 1) * A_BUF_ELEMS, tid);           \
        const _Float16* abase_;                                               \
        int astride_;                                                         \
        if (kbv_ < N_SPLINE_CHUNKS) {                                         \
            abase_   = abuf + (kbv_ & 1) * A_BUF_ELEMS;                       \
            astride_ = K_CHUNK;                                               \
        } else {              /* SiLU chunk: fragments direct from sbuf */    \
            abase_   = sbuf + (k0_ & 255);                                    \
            astride_ = IN_F;                                                  \
        }                                                                     \
        V16U afr_[4];                                                         \
        _Pragma("unroll")                                                     \
        for (int mt = 0; mt < 4; ++mt) {                                      \
            const int m_ = mt * 16 + (lane & 15);                             \
            const _Float16* ap_ = abase_ + m_ * astride_ + 16 * g;            \
            afr_[mt].h[0] = *(const v8h*)(ap_);        /* K = 16g..16g+7 */   \
            afr_[mt].h[1] = *(const v8h*)(ap_ + 8);    /* K = 16g+8..+7  */   \
        }                                                                     \
        _Pragma("unroll")                                                     \
        for (int mt = 0; mt < 4; ++mt)                                        \
            _Pragma("unroll")                                                 \
            for (int nt = 0; nt < 2; ++nt)                                    \
                acc[mt][nt] = __builtin_amdgcn_wmma_f32_16x16x32_f16(         \
                    false, afr_[mt].v, false, BC[nt].v,                       \
                    (short)0, acc[mt][nt], false, false);                     \
    } while (0)

    // Prologue: chunk 0 basis into buffer 0, B fragments for chunk 0.
    buildChunk(0, zbuf, abuf, tid);
    V16U bA[2], bB[2];
    LOADB(0, bA);

    for (int kb = 0; kb < N_CHUNKS; kb += 2) {
        STEP(kb,     bA, bB);
        STEP(kb + 1, bB, bA);
    }
#undef STEP
#undef LOADB

    // ---- Epilogue: C/D layout => M = r + 8g, N = lane&15 ----
    #pragma unroll
    for (int mt = 0; mt < 4; ++mt) {
        #pragma unroll
        for (int nt = 0; nt < 2; ++nt) {
            const int col   = wave * 32 + nt * 16 + (lane & 15);
            const int rbase = rowBase + mt * 16 + 8 * g;
            #pragma unroll
            for (int r = 0; r < 8; ++r)
                out[(size_t)(rbase + r) * OUT_F + col] = acc[mt][nt][r];
        }
    }
}

extern "C" void kernel_launch(void* const* d_in, const int* in_sizes, int n_in,
                              void* d_out, int out_size, void* d_ws, size_t ws_size,
                              hipStream_t stream) {
    const float* x      = (const float*)d_in[0];
    const float* gamma  = (const float*)d_in[1];
    const float* beta   = (const float*)d_in[2];
    const float* coeff  = (const float*)d_in[3];
    const float* base_w = (const float*)d_in[4];
    float* out = (float*)d_out;

    _Float16* Wt = (_Float16*)d_ws;          // 256*3072*2 = 1.5 MB scratch

    pack_weights_kernel<<<(OUT_F * K_TOTAL + 255) / 256, 256, 0, stream>>>(
        coeff, base_w, Wt);

    const size_t shmem = (size_t)M_TILE * IN_F * sizeof(float)        // zbuf
                       + (size_t)M_TILE * IN_F * sizeof(_Float16)     // sbuf
                       + 2u * A_BUF_ELEMS * sizeof(_Float16);         // abuf x2
    (void)hipFuncSetAttribute((const void*)kan_fused_kernel,
                              hipFuncAttributeMaxDynamicSharedMemorySize,
                              (int)shmem);   // CDNA5: up to 320 KB LDS / WGP
    kan_fused_kernel<<<32768 / M_TILE, 256, shmem, stream>>>(
        x, gamma, beta, Wt, out);
}
